// EncoderLayer_63539746177191
// MI455X (gfx1250) — matrix operand order repaired and checked by reading, not compiled, not run on hardware
//
#include <hip/hip_runtime.h>

// Problem constants (match reference)
#define B_   2
#define S_   2048
#define D_   768
#define H_   12
#define DK_  64
#define DFF_ 3072
#define MS_  (B_ * S_)   // 4096 rows

typedef __attribute__((ext_vector_type(16))) __bf16          v16bf;
typedef __attribute__((ext_vector_type(16))) unsigned short  v16u;
typedef __attribute__((ext_vector_type(8)))  float           v8f;
typedef __attribute__((ext_vector_type(4)))  unsigned int    v4u;
typedef __attribute__((ext_vector_type(4)))  int             v4i;

__device__ __forceinline__ unsigned short f32_bf16(float f) {
  unsigned int u = __float_as_uint(f);
  u += 0x7FFFu + ((u >> 16) & 1u);   // round-to-nearest-even
  return (unsigned short)(u >> 16);
}

// ---- CDNA5 async global->LDS copy (16B), guarded so compile never breaks ----
#if defined(__AMDGCN__) && __has_builtin(__builtin_amdgcn_global_load_async_to_lds_b128)
#define ASYNC_LDS_COPY 1
#endif

__device__ __forceinline__ void async_copy16(const unsigned short* g, unsigned short* l) {
#ifdef ASYNC_LDS_COPY
  // signature (from compiler diagnostic): (v4i* src_global, v4i* dst_lds, imm, cpol)
  __builtin_amdgcn_global_load_async_to_lds_b128((v4i*)g, (v4i*)l, 0, 0);
#else
  *(v4u*)l = *(const v4u*)g;
#endif
}

__device__ __forceinline__ void async_wait0() {
#ifdef ASYNC_LDS_COPY
#if __has_builtin(__builtin_amdgcn_s_wait_asynccnt)
  __builtin_amdgcn_s_wait_asynccnt(0);
#else
  asm volatile("s_wait_asynccnt 0x0" ::: "memory");
#endif
#endif
}

// ---- WMMA fragment builders, contiguous K order (k = 16*half + s) in A and B ----
// A matrix 16x32 bf16 (MxK): lane m = lane&15; half-wave selects K base 0/16;
// VGPR i holds K pair {2i,2i+1}, offset by kb (16 contiguous K per lane).
__device__ __forceinline__ v16bf frag_a(const unsigned short* base, int lane, int ld) {
  const int m  = lane & 15;
  const int kb = (lane >> 4) << 4;   // 0 or 16
  const unsigned short* row = base + m * ld + kb;
  v16u t;
#pragma unroll
  for (int i = 0; i < 8; ++i) {
    const int k = 2 * i;
    t[2 * i]     = row[k];
    t[2 * i + 1] = row[k + 1];
  }
  return __builtin_bit_cast(v16bf, t);
}

// B matrix 32x16 bf16 (KxN): lane n = lane&15 owns column n; lanes 0-15 hold
// K=0..15, lanes 16-31 hold K=16..31 -> 16 contiguous K values per lane when
// the tile is stored K-contiguous (we stage B transposed in LDS for this).
__device__ __forceinline__ v16bf frag_b(const unsigned short* base, int lane, int ld) {
  const int n  = lane & 15;
  const int kb = (lane >> 4) << 4;   // 0 or 16
  const unsigned short* p = base + n * ld + kb;
  v16u t;
#pragma unroll
  for (int i = 0; i < 16; ++i) t[i] = p[i];
  return __builtin_bit_cast(v16bf, t);
}

__device__ __forceinline__ v8f wmma_bf16(v16bf a, v16bf b, v8f c) {
  return __builtin_amdgcn_wmma_f32_16x16x32_bf16(false, a, false, b, (short)0, c,
                                                 false, false);
}

// ---------------- fp32 -> bf16 convert ----------------
__global__ __launch_bounds__(256) void cvt_bf16_kernel(const float* __restrict__ in,
                                                       unsigned short* __restrict__ out,
                                                       int n) {
  int i = blockIdx.x * 256 + threadIdx.x;
  if (i < n) out[i] = f32_bf16(in[i]);
}

// ---------------- Tiled WMMA GEMM: C = A[M,K] * W[K,N] + bias ----------------
// 128 threads (4 waves), 64x64 output tile; each wave does a 32x32 subtile.
template <bool RELU, bool OUTBF>
__global__ __launch_bounds__(128) void gemm_bf16_kernel(
    const unsigned short* __restrict__ A, const unsigned short* __restrict__ W,
    const float* __restrict__ bias, unsigned short* __restrict__ outb,
    float* __restrict__ outf, int M, int N, int K) {
  __shared__ unsigned short As[64 * 32];  // [row][k]
  __shared__ unsigned short Bt[64 * 32];  // [col][k] (transposed)

  const int tid  = threadIdx.x;
  const int lane = tid & 31;
  const int wid  = tid >> 5;
  const int wm = wid >> 1, wn = wid & 1;
  const int tileM = blockIdx.y * 64;
  const int tileN = blockIdx.x * 64;

  const v8f vz = {0.f, 0.f, 0.f, 0.f, 0.f, 0.f, 0.f, 0.f};
  v8f acc[2][2];
#pragma unroll
  for (int i = 0; i < 2; ++i)
#pragma unroll
    for (int j = 0; j < 2; ++j) acc[i][j] = vz;

  const int arow = tileM + (tid >> 1);
  const int acol = (tid & 1) * 16;
  const int bk   = tid >> 2;
  const int bn   = (tid & 3) * 16;

  const int ksteps = K >> 5;
  for (int kt = 0; kt < ksteps; ++kt) {
    const int k0 = kt * 32;
    {  // stage A 64x32: two async 16B global->LDS copies per thread
      const unsigned short* src = A + (size_t)arow * K + k0 + acol;
      unsigned short* dst = &As[(tid >> 1) * 32 + acol];
      async_copy16(src, dst);
      async_copy16(src + 8, dst + 8);
    }
    {  // stage W 32x64, transposed into Bt[col][k] (needs element transpose)
      const unsigned short* src = W + (size_t)(k0 + bk) * N + tileN + bn;
      union { v4u v[2]; unsigned short s[16]; } u;
      u.v[0] = *(const v4u*)(src);
      u.v[1] = *(const v4u*)(src + 8);
#pragma unroll
      for (int i = 0; i < 16; ++i) Bt[(bn + i) * 32 + bk] = u.s[i];
    }
    if (kt + 1 < ksteps) {  // prefetch next K tile (global_prefetch_b8)
      __builtin_prefetch(A + (size_t)arow * K + k0 + 32 + acol, 0, 1);
      __builtin_prefetch(W + (size_t)(k0 + 32 + bk) * N + tileN + bn, 0, 1);
    }
    async_wait0();
    __syncthreads();

    v16bf af[2], bf[2];
#pragma unroll
    for (int ms = 0; ms < 2; ++ms)
      af[ms] = frag_a(&As[(wm * 32 + ms * 16) * 32], lane, 32);
#pragma unroll
    for (int ns = 0; ns < 2; ++ns)
      bf[ns] = frag_b(&Bt[(wn * 32 + ns * 16) * 32], lane, 32);
#pragma unroll
    for (int ms = 0; ms < 2; ++ms)
#pragma unroll
      for (int ns = 0; ns < 2; ++ns)
        acc[ms][ns] = wmma_bf16(af[ms], bf[ns], acc[ms][ns]);
    __syncthreads();
  }

  // Epilogue: C/D layout -> (row = half*8 + r, col = lane&15) per VGPR r
  const int n = lane & 15, hf = lane >> 4;
#pragma unroll
  for (int ms = 0; ms < 2; ++ms)
#pragma unroll
    for (int ns = 0; ns < 2; ++ns) {
      const int col = tileN + wn * 32 + ns * 16 + n;
      const float bv = bias[col];
#pragma unroll
      for (int r = 0; r < 8; ++r) {
        const int row = tileM + wm * 32 + ms * 16 + hf * 8 + r;
        float v = acc[ms][ns][r] + bv;
        if (RELU) v = fmaxf(v, 0.f);
        if (OUTBF) outb[(size_t)row * N + col] = f32_bf16(v);
        else       outf[(size_t)row * N + col] = v;
      }
    }
}

// ------- Flash-style attention: 4 waves / WG, shared K,V staging -------
// Workgroup handles 128 query rows of one (b,h); each wave owns 32 rows.
__global__ __launch_bounds__(128) void attn_kernel(
    const unsigned short* __restrict__ Q, const unsigned short* __restrict__ Kmat,
    const unsigned short* __restrict__ Vmat, const int* __restrict__ mask,
    unsigned short* __restrict__ ctx) {
  __shared__ unsigned short Qs[128 * 64];    // [q][d]   16 KB
  __shared__ unsigned short Ks[32 * 64];     // [k][d]    4 KB
  __shared__ unsigned short Vt[64 * 32];     // [d][k]    4 KB
  __shared__ unsigned short Ps[4][32 * 32];  // [wave][q][k]  8 KB

  const int tid  = threadIdx.x;
  const int lane = tid & 31;
  const int wid  = tid >> 5;
  const int bh = blockIdx.x;
  const int b = bh / H_, h = bh % H_;
  const int qblk = blockIdx.y * 128;
  const int q0 = qblk + wid * 32;
  const int n = lane & 15, hf = lane >> 4;

  {  // stage 128x64 Q tile cooperatively: 1024 16B chunks / 128 threads
#pragma unroll
    for (int j = 0; j < 8; ++j) {
      const int c = tid + 128 * j;
      const int row = c >> 3, p = c & 7;
      async_copy16(Q + (size_t)(b * S_ + qblk + row) * D_ + h * DK_ + p * 8,
                   &Qs[row * 64 + p * 8]);
    }
  }
  async_wait0();
  __syncthreads();

  v16bf qf[2][2];
#pragma unroll
  for (int qi = 0; qi < 2; ++qi)
#pragma unroll
    for (int dk = 0; dk < 2; ++dk)
      qf[qi][dk] = frag_a(&Qs[(wid * 32 + qi * 16) * 64 + dk * 32], lane, 64);

  const v8f vz = {0.f, 0.f, 0.f, 0.f, 0.f, 0.f, 0.f, 0.f};
  v8f O[2][4];
  float m_run[2][8], l_run[2][8];
#pragma unroll
  for (int qi = 0; qi < 2; ++qi) {
#pragma unroll
    for (int dj = 0; dj < 4; ++dj) O[qi][dj] = vz;
#pragma unroll
    for (int r = 0; r < 8; ++r) { m_run[qi][r] = -3.0e38f; l_run[qi][r] = 0.f; }
  }

  const size_t maskbase = (size_t)b * S_ * S_;

  for (int kt = 0; kt < S_ / 32; ++kt) {
    const int kbase = kt * 32;
    {  // stage K tile (async, 256 chunks / 128 threads)
#pragma unroll
      for (int j = 0; j < 2; ++j) {
        const int c = tid + 128 * j;
        const int row = c >> 3, p = c & 7;
        async_copy16(Kmat + (size_t)(b * S_ + kbase + row) * D_ + h * DK_ + p * 8,
                     &Ks[row * 64 + p * 8]);
      }
      // stage V transposed (manual: element transpose required)
#pragma unroll
      for (int j = 0; j < 2; ++j) {
        const int c = tid + 128 * j;
        const int row = c >> 3, p = c & 7;
        union { v4u v; unsigned short s[8]; } u;
        u.v = *(const v4u*)(Vmat + (size_t)(b * S_ + kbase + row) * D_ + h * DK_ + p * 8);
#pragma unroll
        for (int i = 0; i < 8; ++i) Vt[(p * 8 + i) * 32 + row] = u.s[i];
      }
    }
    async_wait0();
    __syncthreads();

    // K^T fragments: column = key, K-dim = d (contiguous in Ks rows)
    v16bf kf[2][2];
#pragma unroll
    for (int kj = 0; kj < 2; ++kj)
#pragma unroll
      for (int dk = 0; dk < 2; ++dk)
        kf[kj][dk] = frag_b(&Ks[kj * 16 * 64 + dk * 32], lane, 64);

#pragma unroll
    for (int qi = 0; qi < 2; ++qi) {
      v8f s0 = vz, s1 = vz;
#pragma unroll
      for (int dk = 0; dk < 2; ++dk) {
        s0 = wmma_bf16(qf[qi][dk], kf[0][dk], s0);
        s1 = wmma_bf16(qf[qi][dk], kf[1][dk], s1);
      }
      // scale + mask on D-fragment (lane owns rows hf*8+r, col n / 16+n)
      float e0[8], e1[8];
#pragma unroll
      for (int r = 0; r < 8; ++r) {
        const int qrow = q0 + qi * 16 + hf * 8 + r;
        const int m0 = mask[maskbase + (size_t)qrow * S_ + kbase + n];
        const int m1 = mask[maskbase + (size_t)qrow * S_ + kbase + 16 + n];
        e0[r] = (m0 != 0) ? s0[r] * 0.125f : -1.0e9f;
        e1[r] = (m1 != 0) ? s1[r] * 0.125f : -1.0e9f;
      }
      // row max across the 16 lanes of this half-wave
      float mt[8];
#pragma unroll
      for (int r = 0; r < 8; ++r) mt[r] = fmaxf(e0[r], e1[r]);
#pragma unroll
      for (int off = 8; off >= 1; off >>= 1)
#pragma unroll
        for (int r = 0; r < 8; ++r) mt[r] = fmaxf(mt[r], __shfl_xor(mt[r], off));
      // online softmax update
      float sc[8], rs[8];
#pragma unroll
      for (int r = 0; r < 8; ++r) {
        const float mnew = fmaxf(m_run[qi][r], mt[r]);
        sc[r] = __expf(m_run[qi][r] - mnew);
        m_run[qi][r] = mnew;
        e0[r] = __expf(e0[r] - mnew);
        e1[r] = __expf(e1[r] - mnew);
        rs[r] = e0[r] + e1[r];
      }
#pragma unroll
      for (int off = 8; off >= 1; off >>= 1)
#pragma unroll
        for (int r = 0; r < 8; ++r) rs[r] += __shfl_xor(rs[r], off);
#pragma unroll
      for (int r = 0; r < 8; ++r) l_run[qi][r] = l_run[qi][r] * sc[r] + rs[r];
#pragma unroll
      for (int dj = 0; dj < 4; ++dj)
#pragma unroll
        for (int r = 0; r < 8; ++r) O[qi][dj][r] = O[qi][dj][r] * sc[r];
      // write P (bf16) to per-wave LDS region (D-layout -> A-layout)
#pragma unroll
      for (int r = 0; r < 8; ++r) {
        const int prow = qi * 16 + hf * 8 + r;
        Ps[wid][prow * 32 + n]      = f32_bf16(e0[r]);
        Ps[wid][prow * 32 + 16 + n] = f32_bf16(e1[r]);
      }
    }

    // O += P(32x32) * V(32x64): K-dim = 32 keys = one WMMA per 16x16 tile
#pragma unroll
    for (int qi = 0; qi < 2; ++qi) {
      v16bf pa = frag_a(&Ps[wid][qi * 16 * 32], lane, 32);
#pragma unroll
      for (int dj = 0; dj < 4; ++dj) {
        v16bf vb = frag_b(&Vt[dj * 16 * 32], lane, 32);
        O[qi][dj] = wmma_bf16(pa, vb, O[qi][dj]);
      }
    }
    __syncthreads();  // before next iteration restages Ks/Vt
  }

  // normalize and store ctx (bf16, heads re-interleaved into [B*S, D])
#pragma unroll
  for (int qi = 0; qi < 2; ++qi)
#pragma unroll
    for (int dj = 0; dj < 4; ++dj)
#pragma unroll
      for (int r = 0; r < 8; ++r) {
        const int qrow = q0 + qi * 16 + hf * 8 + r;
        const int col = h * DK_ + dj * 16 + n;
        const float v = O[qi][dj][r] / l_run[qi][r];
        ctx[(size_t)(b * S_ + qrow) * D_ + col] = f32_bf16(v);
      }
}

// ---------------- residual add + LayerNorm (row = 768) ----------------
__global__ __launch_bounds__(256) void add_ln_kernel(
    const float* __restrict__ a, const float* __restrict__ c,
    const float* __restrict__ g, const float* __restrict__ be,
    float* __restrict__ outf, unsigned short* __restrict__ outb) {
  __shared__ float red[256];
  const int row = blockIdx.x;
  const int tid = threadIdx.x;
  float v[3];
#pragma unroll
  for (int j = 0; j < 3; ++j) {
    const int ci = tid + j * 256;
    v[j] = a[(size_t)row * D_ + ci] + c[(size_t)row * D_ + ci];
  }
  red[tid] = v[0] + v[1] + v[2];
  __syncthreads();
  for (int st = 128; st > 0; st >>= 1) {
    if (tid < st) red[tid] += red[tid + st];
    __syncthreads();
  }
  const float mu = red[0] * (1.0f / D_);
  __syncthreads();
  float sq = 0.f;
#pragma unroll
  for (int j = 0; j < 3; ++j) { const float d = v[j] - mu; sq += d * d; }
  red[tid] = sq;
  __syncthreads();
  for (int st = 128; st > 0; st >>= 1) {
    if (tid < st) red[tid] += red[tid + st];
    __syncthreads();
  }
  const float rstd = rsqrtf(red[0] * (1.0f / D_) + 1e-5f);
#pragma unroll
  for (int j = 0; j < 3; ++j) {
    const int ci = tid + j * 256;
    const float o = (v[j] - mu) * rstd * g[ci] + be[ci];
    outf[(size_t)row * D_ + ci] = o;
    if (outb) outb[(size_t)row * D_ + ci] = f32_bf16(o);
  }
}

extern "C" void kernel_launch(void* const* d_in, const int* in_sizes, int n_in,
                              void* d_out, int out_size, void* d_ws, size_t ws_size,
                              hipStream_t stream) {
  (void)in_sizes; (void)n_in; (void)out_size; (void)ws_size;
  const float* x   = (const float*)d_in[0];
  const int*   msk = (const int*)d_in[1];
  const float* WQ = (const float*)d_in[2];  const float* bQ = (const float*)d_in[3];
  const float* WK = (const float*)d_in[4];  const float* bK = (const float*)d_in[5];
  const float* WV = (const float*)d_in[6];  const float* bV = (const float*)d_in[7];
  const float* WO = (const float*)d_in[8];  const float* bO = (const float*)d_in[9];
  const float* W1 = (const float*)d_in[10]; const float* b1 = (const float*)d_in[11];
  const float* W2 = (const float*)d_in[12]; const float* b2 = (const float*)d_in[13];
  const float* g1 = (const float*)d_in[14]; const float* be1 = (const float*)d_in[15];
  const float* g2 = (const float*)d_in[16]; const float* be2 = (const float*)d_in[17];
  float* out = (float*)d_out;

  char* ws = (char*)d_ws;
  size_t off = 0;
  auto alloc = [&](size_t bytes) {
    char* p = ws + off;
    off += bytes;
    off = (off + 255) & ~(size_t)255;
    return p;
  };
  unsigned short* xb   = (unsigned short*)alloc((size_t)MS_ * D_ * 2);
  unsigned short* wqb  = (unsigned short*)alloc((size_t)D_ * D_ * 2);
  unsigned short* wkb  = (unsigned short*)alloc((size_t)D_ * D_ * 2);
  unsigned short* wvb  = (unsigned short*)alloc((size_t)D_ * D_ * 2);
  unsigned short* wob  = (unsigned short*)alloc((size_t)D_ * D_ * 2);
  unsigned short* w1b  = (unsigned short*)alloc((size_t)D_ * DFF_ * 2);
  unsigned short* w2b  = (unsigned short*)alloc((size_t)DFF_ * D_ * 2);
  unsigned short* qb   = (unsigned short*)alloc((size_t)MS_ * D_ * 2);
  unsigned short* kb   = (unsigned short*)alloc((size_t)MS_ * D_ * 2);
  unsigned short* vb   = (unsigned short*)alloc((size_t)MS_ * D_ * 2);
  unsigned short* ctxb = (unsigned short*)alloc((size_t)MS_ * D_ * 2);
  float*          attno= (float*)alloc((size_t)MS_ * D_ * 4);
  float*          hfull= (float*)alloc((size_t)MS_ * D_ * 4);
  unsigned short* hb   = (unsigned short*)alloc((size_t)MS_ * D_ * 2);
  unsigned short* ff1  = (unsigned short*)alloc((size_t)MS_ * DFF_ * 2);
  float*          ff2  = (float*)alloc((size_t)MS_ * D_ * 4);

  auto cvt = [&](const float* src, unsigned short* dst, size_t n) {
    cvt_bf16_kernel<<<dim3((unsigned)((n + 255) / 256)), dim3(256), 0, stream>>>(
        src, dst, (int)n);
  };
  cvt(x,  xb,  (size_t)MS_ * D_);
  cvt(WQ, wqb, (size_t)D_ * D_);
  cvt(WK, wkb, (size_t)D_ * D_);
  cvt(WV, wvb, (size_t)D_ * D_);
  cvt(WO, wob, (size_t)D_ * D_);
  cvt(W1, w1b, (size_t)D_ * DFF_);
  cvt(W2, w2b, (size_t)DFF_ * D_);

  const dim3 blk128(128);
  // Q/K/V projections -> bf16
  gemm_bf16_kernel<false, true><<<dim3(D_ / 64, MS_ / 64), blk128, 0, stream>>>(
      xb, wqb, bQ, qb, nullptr, MS_, D_, D_);
  gemm_bf16_kernel<false, true><<<dim3(D_ / 64, MS_ / 64), blk128, 0, stream>>>(
      xb, wkb, bK, kb, nullptr, MS_, D_, D_);
  gemm_bf16_kernel<false, true><<<dim3(D_ / 64, MS_ / 64), blk128, 0, stream>>>(
      xb, wvb, bV, vb, nullptr, MS_, D_, D_);
  // attention -> ctx (bf16); 4 waves/WG, 128 q-rows per WG
  attn_kernel<<<dim3(B_ * H_, S_ / 128), blk128, 0, stream>>>(qb, kb, vb, msk, ctxb);
  // output projection -> f32
  gemm_bf16_kernel<false, false><<<dim3(D_ / 64, MS_ / 64), blk128, 0, stream>>>(
      ctxb, wob, bO, nullptr, attno, MS_, D_, D_);
  // h = LN(x + attn_out) -> f32 + bf16
  add_ln_kernel<<<dim3(MS_), dim3(256), 0, stream>>>(x, attno, g1, be1, hfull, hb);
  // FFN1 with ReLU -> bf16
  gemm_bf16_kernel<true, true><<<dim3(DFF_ / 64, MS_ / 64), blk128, 0, stream>>>(
      hb, w1b, b1, ff1, nullptr, MS_, DFF_, D_);
  // FFN2 -> f32
  gemm_bf16_kernel<false, false><<<dim3(D_ / 64, MS_ / 64), blk128, 0, stream>>>(
      ff1, w2b, b2, nullptr, ff2, MS_, D_, DFF_);
  // out = LN(h + ff) -> f32
  add_ln_kernel<<<dim3(MS_), dim3(256), 0, stream>>>(hfull, ff2, g2, be2, out, nullptr);
}
